// CNN_2233382994624
// MI455X (gfx1250) — matrix operand order repaired and checked by reading, not compiled, not run on hardware
//
#include <hip/hip_runtime.h>
#include <hip/hip_bf16.h>

// ---------------------------------------------------------------------------
// CDNA5 (gfx1250) SE(3)-CNN forward.
// bf16 activations/weights, fp32 accumulation via v_wmma_f32_16x16x32_bf16.
// Weight tiles double-buffered in LDS via TDM (tensor_load_to_lds),
// A-fragments read back as 16-byte LDS vector loads.
// ---------------------------------------------------------------------------

typedef __bf16 bf16_t;
typedef __attribute__((ext_vector_type(16))) __bf16 v16bf;
typedef __attribute__((ext_vector_type(8)))  __bf16 v8bf;
typedef __attribute__((ext_vector_type(8)))  float  v8f;
typedef __attribute__((ext_vector_type(4)))  unsigned int u32x4;
typedef __attribute__((ext_vector_type(8)))  int i32x8;
typedef __attribute__((ext_vector_type(4)))  int i32x4;

#define EPSV 1e-5f
#define NB 32            // batch
#define KCHUNK 128       // K elements staged per TDM shot (16x128 bf16 = 4 KB)

#if defined(__gfx1250__) && defined(__has_builtin)
# if __has_builtin(__builtin_amdgcn_tensor_load_to_lds)
#  define HAVE_TDM 1
# endif
#endif
#if __has_include(<hip/amd_detail/amd_gfx1250_TDM.h>)
# define TDM_6ARG 1
#endif

__device__ __forceinline__ float bf2f(bf16_t b) {
    union { unsigned short s; bf16_t b; } u; u.b = b;
    union { unsigned u; float f; } v; v.u = ((unsigned)u.s) << 16;
    return v.f;
}
__device__ __forceinline__ bf16_t f2bf(float f) {
    union { unsigned u; float f; } v; v.f = f;
    unsigned r = (v.u + 0x7FFFu + ((v.u >> 16) & 1u)) >> 16;
    union { unsigned short s; bf16_t b; } u; u.s = (unsigned short)r;
    return u.b;
}

// --------------------------- weight fp32 -> bf16 ---------------------------
__global__ void k_cvt_w(const float* __restrict__ src, bf16_t* __restrict__ dst, long n) {
    for (long i = (long)blockIdx.x * blockDim.x + threadIdx.x; i < n;
         i += (long)gridDim.x * blockDim.x)
        dst[i] = f2bf(src[i]);
}

// --------------------------- input mean/var --------------------------------
__global__ void k_in_stats(const float* __restrict__ x, float* __restrict__ stats, long n) {
    __shared__ float ss[2];
    if (threadIdx.x == 0) { ss[0] = 0.f; ss[1] = 0.f; }
    __syncthreads();
    float s0 = 0.f, s1 = 0.f;
    for (long i = (long)blockIdx.x * blockDim.x + threadIdx.x; i < n;
         i += (long)gridDim.x * blockDim.x) {
        float v = x[i]; s0 += v; s1 += v * v;
    }
    atomicAdd(&ss[0], s0); atomicAdd(&ss[1], s1);
    __syncthreads();
    if (threadIdx.x == 0) { atomicAdd(&stats[0], ss[0]); atomicAdd(&stats[1], ss[1]); }
}

__global__ void k_in_norm(const float* __restrict__ x, bf16_t* __restrict__ out,
                          const float* __restrict__ stats, long n) {
    float mu  = stats[0] / (float)n;
    float var = stats[1] / (float)n - mu * mu;
    float inv = rsqrtf(var + EPSV);
    for (long i = (long)blockIdx.x * blockDim.x + threadIdx.x; i < n;
         i += (long)gridDim.x * blockDim.x)
        out[i] = f2bf((x[i] - mu) * inv);
}

// --------------------------- TDM weight stage ------------------------------
#if defined(HAVE_TDM)
__device__ __forceinline__ void tdm_stage_weights(const bf16_t* gsrc, unsigned ldsAddr,
                                                  int remK, int remM, int Ktot) {
    unsigned long long gaddr = (unsigned long long)(const void*)gsrc;
    unsigned td0 = (unsigned)remK;                 // tensor_dim0 (OOB cols -> 0)
    unsigned td1 = (unsigned)remM;                 // tensor_dim1 (OOB rows -> 0)
    unsigned long long st0 = (unsigned long long)Ktot;

    u32x4 g0;
    g0[0] = 1u;                                    // count=1, user mode
    g0[1] = ldsAddr;                               // lds_addr (bytes)
    g0[2] = (unsigned)(gaddr & 0xFFFFFFFFull);     // global_addr[31:0]
    g0[3] = (unsigned)((gaddr >> 32) & 0x01FFFFFFull) | 0x80000000u;  // +type=2

    i32x8 g1;
    g1[0] = (int)(1u << 16);                       // data_size=1 (2 bytes/elem)
    g1[1] = (int)((td0 & 0xFFFFu) << 16);          // tensor_dim0[15:0]
    g1[2] = (int)(((td0 >> 16) & 0xFFFFu) | ((td1 & 0xFFFFu) << 16));
    g1[3] = (int)(((td1 >> 16) & 0xFFFFu) | ((unsigned)KCHUNK << 16)); // tile_dim0
    g1[4] = (int)(16u);                            // tile_dim1=16, tile_dim2=0
    g1[5] = (int)(st0 & 0xFFFFFFFFull);            // dim0_stride[31:0]
    g1[6] = (int)((st0 >> 32) & 0xFFFFull);        // dim0_stride[47:32]
    g1[7] = 0;

    i32x4 gz = {};
#if defined(TDM_6ARG)
    i32x8 gz8 = {};
    __builtin_amdgcn_tensor_load_to_lds(g0, g1, gz, gz, gz8, 0);
#else
    __builtin_amdgcn_tensor_load_to_lds(g0, g1, gz, gz, 0);
#endif
}
#endif

// --------------------------- per-chunk compute -----------------------------
__device__ __forceinline__ void conv_compute_chunk(
    const bf16_t* __restrict__ curw, const bf16_t* __restrict__ in,
    v8f& acc, int kk, int Ktot, bool nvalid,
    int b, int cin, int Din, int id0, int ih0, int iw0, int col, int hi,
    bf16_t BZ) {
#pragma unroll
    for (int q = 0; q < KCHUNK / 32; ++q) {
        const int kb = q * 32;

        // ---- A fragment: two contiguous 8xbf16 runs per lane (ds_load_b128) ----
        const bf16_t* ar = curw + col * KCHUNK + kb + hi * 8;
        v8bf alo = *(const v8bf*)ar;           // K = kb+hi*8 + [0..7]
        v8bf ahi = *(const v8bf*)(ar + 16);    // K = kb+hi*8+16 + [0..7]
        v16bf a;
#pragma unroll
        for (int j = 0; j < 8; ++j) { a[j] = alo[j]; a[j + 8] = ahi[j]; }

        // ---- B fragment: 32x16 bf16 im2col patch from global ----
        v16bf bm;
#pragma unroll
        for (int j = 0; j < 16; ++j) {
            int k = kk + kb + j + hi * 8 + ((j >> 3) << 3);
            bf16_t val = BZ;
            if (nvalid && k < Ktot) {
                int ci = k / 343; int t = k - ci * 343;
                int kd = t / 49;  t -= kd * 49;
                int kh = t / 7;   int kw = t - kh * 7;
                int id = id0 + kd, ih = ih0 + kh, iw = iw0 + kw;
                if ((unsigned)id < (unsigned)Din &&
                    (unsigned)ih < (unsigned)Din &&
                    (unsigned)iw < (unsigned)Din) {
                    val = in[((((long)b * cin + ci) * Din + id) * Din + ih) * (long)Din + iw];
                }
            }
            bm[j] = val;
        }

        acc = __builtin_amdgcn_wmma_f32_16x16x32_bf16(
            false, a, false, bm, (short)0, acc, false, false);
    }
}

// --------------------------- implicit-GEMM conv (WMMA) ---------------------
// Weights [cout][cin*343] bf16.  One 16-row M tile per block, shared by 8
// waves (each wave owns a 16-wide N subtile).  16x128 weight tiles are
// double-buffered in LDS by the TDM; one barrier per 4 WMMAs.
__global__ void __launch_bounds__(256)
k_conv_wmma(const bf16_t* __restrict__ in, const bf16_t* __restrict__ w,
            bf16_t* __restrict__ out,
            int cin, int cout, int Din, int Dout, int stride,
            int outChAlloc, int Ktot, int Ntot) {
    __shared__ __align__(16) bf16_t lds_w[2 * 16 * KCHUNK];   // 8 KB, 2 stages

    const int lane = threadIdx.x & 31;
    const int wave = threadIdx.x >> 5;
    const int hi   = (lane & 16) ? 1 : 0;      // upper half-wave
    const int col  = lane & 15;

    const int S2 = Dout * Dout;
    const int S3 = Dout * S2;

    const int nBase = (blockIdx.x * 8 + wave) * 16;
    const int mBase = blockIdx.y * 16;

    // decompose this lane's output position (B column)
    const int n = nBase + col;
    const bool nvalid = (n < Ntot);
    int b = 0, od = 0, oh = 0, ow = 0;
    if (nvalid) {
        b = n / S3; int r = n - b * S3;
        od = r / S2; r -= od * S2;
        oh = r / Dout; ow = r - oh * Dout;
    }
    const int id0 = od * stride - 3;
    const int ih0 = oh * stride - 3;
    const int iw0 = ow * stride - 3;

    const bf16_t BZ = f2bf(0.f);
    const bf16_t* wbase = w + (long)mBase * Ktot;
    const int nChunks = (Ktot + KCHUNK - 1) / KCHUNK;
    v8f acc = {};

#if defined(HAVE_TDM)
    if (wave == 0)
        tdm_stage_weights(wbase, (unsigned)(unsigned long long)(void*)&lds_w[0],
                          Ktot, cout - mBase, Ktot);

    for (int c = 0; c < nChunks; ++c) {
        const int kk = c * KCHUNK;
        bf16_t* cur = &lds_w[(c & 1) * 16 * KCHUNK];
        bf16_t* nxt = &lds_w[((c + 1) & 1) * 16 * KCHUNK];

        if (wave == 0) __builtin_amdgcn_s_wait_tensorcnt(0);
        __syncthreads();                        // publish stage c; stage c+1 LDS free

        if (wave == 0 && c + 1 < nChunks) {
            __builtin_prefetch(wbase + kk + 2 * KCHUNK, 0, 1);   // global_prefetch
            tdm_stage_weights(wbase + (kk + KCHUNK),
                              (unsigned)(unsigned long long)(void*)nxt,
                              Ktot - (kk + KCHUNK), cout - mBase, Ktot);
        }

        conv_compute_chunk(cur, in, acc, kk, Ktot, nvalid,
                           b, cin, Din, id0, ih0, iw0, col, hi, BZ);
    }
#else
    for (int c = 0; c < nChunks; ++c) {
        const int kk = c * KCHUNK;
        bf16_t* cur = &lds_w[0];
        for (int t = threadIdx.x; t < 16 * KCHUNK; t += 256) {
            int r = t / KCHUNK, cc = t % KCHUNK;
            int gm = mBase + r, gk = kk + cc;
            cur[t] = (gm < cout && gk < Ktot) ? w[(long)gm * Ktot + gk] : BZ;
        }
        __syncthreads();
        conv_compute_chunk(cur, in, acc, kk, Ktot, nvalid,
                           b, cin, Din, id0, ih0, iw0, col, hi, BZ);
        __syncthreads();
    }
#endif

    // ---- store D ----
    if (nvalid) {
        long spOff = ((long)od * Dout + oh) * Dout + ow;
#pragma unroll
        for (int r = 0; r < 8; ++r) {
            int mo = mBase + r + hi * 8;
            if (mo < cout)
                out[((long)b * outChAlloc + mo) * S3 + spOff] = f2bf(acc[r]);
        }
    }
}

// --------------------------- SE(3) batch-norm ------------------------------
__global__ void k_bn_stats(const bf16_t* __restrict__ buf, float* __restrict__ stats,
                           int s, int v, long S, int chAlloc) {
    __shared__ float part[32];
    const int nfeat = s + v;
    for (int i = threadIdx.x; i < nfeat; i += blockDim.x) part[i] = 0.f;
    __syncthreads();
    const int cout = s + 3 * v;
    const long total = (long)NB * cout * S;
    for (long idx = (long)blockIdx.x * blockDim.x + threadIdx.x; idx < total;
         idx += (long)gridDim.x * blockDim.x) {
        long sp = idx % S; long t = idx / S;
        int ch = (int)(t % cout); int b = (int)(t / cout);
        float x = bf2f(buf[((long)b * chAlloc + ch) * S + sp]);
        int f = (ch < s) ? ch : s + (ch - s) / 3;
        atomicAdd(&part[f], x * x);          // ds_add_f32
    }
    __syncthreads();
    for (int i = threadIdx.x; i < nfeat; i += blockDim.x)
        atomicAdd(&stats[i], part[i]);
}

__global__ void k_bn_apply(bf16_t* __restrict__ buf, const float* __restrict__ stats,
                           const float* __restrict__ bnw, const float* __restrict__ bias,
                           int s, int v, long S, int chAlloc, int doBias) {
    const int cout = s + 3 * v;
    const long total = (long)NB * cout * S;
    const float scnt = (float)((long)NB * S);
    for (long idx = (long)blockIdx.x * blockDim.x + threadIdx.x; idx < total;
         idx += (long)gridDim.x * blockDim.x) {
        long sp = idx % S; long t = idx / S;
        int ch = (int)(t % cout); int b = (int)(t / cout);
        int f = (ch < s) ? ch : s + (ch - s) / 3;
        float cnt = (ch < s) ? scnt : 3.f * scnt;
        float fac = bnw[f] * rsqrtf(stats[f] / cnt + EPSV);
        long off = ((long)b * chAlloc + ch) * S + sp;
        float x = bf2f(buf[off]) * fac;
        if (ch < s && doBias) { x += bias[ch]; x = x > 0.f ? x : 0.f; }
        buf[off] = f2bf(x);
    }
}

// --------------------------- tensor square ---------------------------------
__global__ void k_tensor_sq(bf16_t* __restrict__ buf, int s, int v, long S, int chAlloc) {
    const long total = (long)NB * 9 * v * S;
    for (long idx = (long)blockIdx.x * blockDim.x + threadIdx.x; idx < total;
         idx += (long)gridDim.x * blockDim.x) {
        long sp = idx % S; long t = idx / S;
        int q = (int)(t % (9 * v)); int b = (int)(t / (9 * v));
        int a = q / 9; int ij = q % 9; int i = ij / 3; int j = ij % 3;
        float vi = bf2f(buf[((long)b * chAlloc + s + 3 * a + i) * S + sp]);
        float vj = bf2f(buf[((long)b * chAlloc + s + 3 * a + j) * S + sp]);
        buf[((long)b * chAlloc + s + 3 * v + q) * S + sp] = f2bf(vi * vj);
    }
}

// --------------------------- final pooling + layernorm ---------------------
__global__ void k_spatial_mean(const bf16_t* __restrict__ buf, float* __restrict__ y) {
    int t = blockIdx.x * blockDim.x + threadIdx.x;   // 0..319 -> (b,c)
    if (t < NB * 10) {
        int b = t / 10, c = t % 10;
        float sum = 0.f;
        for (int i = 0; i < 64; ++i) sum += bf2f(buf[((long)b * 10 + c) * 64 + i]);
        y[t] = sum * (1.f / 64.f);
    }
}

__global__ void k_final_ln(const float* __restrict__ y, const float* __restrict__ gamma,
                           const float* __restrict__ beta, float* __restrict__ out) {
    __shared__ float mu[10], iv[10];
    int t = threadIdx.x;
    if (t < 10) {
        float s = 0.f, sq = 0.f;
        for (int b = 0; b < NB; ++b) { float v = y[b * 10 + t]; s += v; sq += v * v; }
        float m = s / (float)NB;
        mu[t] = m;
        iv[t] = rsqrtf(sq / (float)NB - m * m + EPSV);
    }
    __syncthreads();
    if (t < NB * 10) {
        int c = t % 10;
        out[t] = (y[t] - mu[c]) * iv[c] * gamma[c] + beta[c];
    }
}

// ---------------------------------------------------------------------------
// Host driver
// ---------------------------------------------------------------------------
struct LCfg { int s_in, v_in, s_out, v_out, stride, Din, Dout; long woff; };

extern "C" void kernel_launch(void* const* d_in, const int* in_sizes, int n_in,
                              void* d_out, int out_size, void* d_ws, size_t ws_size,
                              hipStream_t stream) {
    (void)in_sizes; (void)n_in; (void)out_size; (void)ws_size;

    static const LCfg L[7] = {
        { 1, 0,  4, 2, 2, 64, 32,       0 },
        { 4, 2, 10, 5, 1, 32, 32,    3430 },
        {10, 5, 10, 5, 2, 32, 16,  243530 },
        {10, 5, 12, 6, 1, 16, 16,  843780 },
        {12, 6, 12, 6, 2, 16,  8, 1564080 },
        {12, 6, 14, 7, 1,  8,  8, 2428440 },
        {14, 7, 10, 0, 2,  8,  4, 3436860 },
    };

    char* ws = (char*)d_ws;
    const size_t OFF_BUF0  = 0;                       // 160 MB (peak: L2 input, 147 MB)
    const size_t OFF_BUF1  = (size_t)160 << 20;       // 64 MB  (peak: L1 input, 59 MB)
    const size_t OFF_WBF   = (size_t)224 << 20;       // 8 MB bf16 weights
    const size_t OFF_STATS = (size_t)232 << 20;       // 64 floats
    const size_t OFF_Y     = OFF_STATS + 4096;        // 320 floats

    bf16_t* buf0  = (bf16_t*)(ws + OFF_BUF0);
    bf16_t* buf1  = (bf16_t*)(ws + OFF_BUF1);
    bf16_t* wbf   = (bf16_t*)(ws + OFF_WBF);
    float*  stats = (float*) (ws + OFF_STATS);
    float*  yvec  = (float*) (ws + OFF_Y);

    const float* x     = (const float*)d_in[0];
    const float* gamma = (const float*)d_in[21];
    const float* beta  = (const float*)d_in[22];
    float* out = (float*)d_out;

    // ---- convert all weights to bf16 ----
    for (int i = 0; i < 7; ++i) {
        int cin  = L[i].s_in + 3 * L[i].v_in + 9 * L[i].v_in;
        int cout = L[i].s_out + 3 * L[i].v_out;
        long n = (long)cout * cin * 343;
        int g = (int)((n + 255) / 256); if (g > 2048) g = 2048;
        k_cvt_w<<<g, 256, 0, stream>>>((const float*)d_in[1 + i], wbf + L[i].woff, n);
    }

    // ---- input normalization (single channel) ----
    const long NIN = (long)NB * 64 * 64 * 64;
    hipMemsetAsync(stats, 0, 64 * sizeof(float), stream);
    k_in_stats<<<2048, 256, 0, stream>>>(x, stats, NIN);
    k_in_norm <<<2048, 256, 0, stream>>>(x, buf0, stats, NIN);

    // ---- 7 layers ----
    for (int i = 0; i < 7; ++i) {
        const int cin   = L[i].s_in + 3 * L[i].v_in + 9 * L[i].v_in;
        const int cout  = L[i].s_out + 3 * L[i].v_out;
        const int Din   = L[i].Din, Dout = L[i].Dout;
        const int Ktot  = cin * 343;
        const int Ntot  = NB * Dout * Dout * Dout;
        const long S    = (long)Dout * Dout * Dout;
        const int chOut = (i < 6) ? (L[i + 1].s_in + 3 * L[i + 1].v_in + 9 * L[i + 1].v_in)
                                  : 10;
        bf16_t* bin  = (i & 1) ? buf1 : buf0;
        bf16_t* bout = (i & 1) ? buf0 : buf1;

        dim3 grid((Ntot + 127) / 128, (cout + 15) / 16);
        k_conv_wmma<<<grid, 256, 0, stream>>>(bin, wbf + L[i].woff, bout,
                                              cin, cout, Din, Dout, L[i].stride,
                                              chOut, Ktot, Ntot);

        long tot = (long)NB * cout * S;
        int g = (int)((tot + 255) / 256); if (g > 4096) g = 4096;
        hipMemsetAsync(stats, 0, 64 * sizeof(float), stream);
        k_bn_stats<<<g, 256, 0, stream>>>(bout, stats, L[i].s_out, L[i].v_out, S, chOut);
        k_bn_apply<<<g, 256, 0, stream>>>(bout, stats,
                                          (const float*)d_in[8 + i],
                                          (i < 6) ? (const float*)d_in[15 + i]
                                                  : (const float*)d_in[15],
                                          L[i].s_out, L[i].v_out, S, chOut, (i < 6) ? 1 : 0);

        if (i < 6 && L[i].v_out > 0) {
            long tsq = (long)NB * 9 * L[i].v_out * S;
            int g2 = (int)((tsq + 255) / 256); if (g2 > 4096) g2 = 4096;
            k_tensor_sq<<<g2, 256, 0, stream>>>(bout, L[i].s_out, L[i].v_out, S, chOut);
        }
    }

    // ---- final: spatial mean + batch layernorm ----
    k_spatial_mean<<<2, 256, 0, stream>>>(buf1, yvec);
    k_final_ln   <<<1, 320, 0, stream>>>(yvec, gamma, beta, out);
}
